// CausalAttention_36996848288146
// MI455X (gfx1250) — matrix operand order repaired and checked by reading, not compiled, not run on hardware
//
#include <hip/hip_runtime.h>
#include <hip/hip_bf16.h>
#include <stdint.h>

#define D_MODEL 2048
#define N_HEADS 16
#define D_HEAD  128
#define BATCH   4
#define SEQ     2048
#define MTOT    (BATCH*SEQ)   // 8192 rows

typedef __attribute__((ext_vector_type(16))) unsigned short v16us;
typedef __attribute__((ext_vector_type(8)))  unsigned short v8us;
typedef __attribute__((ext_vector_type(16))) __bf16        v16bf;
typedef __attribute__((ext_vector_type(8)))  float         v8f;

union FragU { v16us v; v8us h[2]; };

__device__ __forceinline__ float bf2f(unsigned short u) {
  union { unsigned u; float f; } w; w.u = ((unsigned)u) << 16; return w.f;
}
__device__ __forceinline__ unsigned short f2bf(float f) {
  union { float f; unsigned u; } w; w.f = f;
  unsigned r = w.u + 0x7FFFu + ((w.u >> 16) & 1u);
  return (unsigned short)(r >> 16);
}

// ---------------------------------------------------------------- convert (x4 vectorized)
__global__ void k_f32_to_bf16(const float* __restrict__ src,
                              unsigned short* __restrict__ dst, int n4) {
  int i = blockIdx.x * blockDim.x + threadIdx.x;
  if (i < n4) {
    float4 v = ((const float4*)src)[i];
    ushort4 o;
    o.x = f2bf(v.x); o.y = f2bf(v.y); o.z = f2bf(v.z); o.w = f2bf(v.w);
    ((ushort4*)dst)[i] = o;
  }
}

// ---------------------------------------------------------------- GEMM
// Y[m][n] = sum_k X[m][k] * W[n][k]   (X: MTOTx2048 bf16, W: 2048x2048 bf16)
// mode 0: Q raw -> [b][h][t][d] bf16     mode 1: K raw -> [b][h][t][d] bf16
// mode 2: V     -> [b][h][d][t] bf16     mode 3: f32 out [m][n]
// Block 128x128, 8 waves (4M x 2N, wave tile 32x64), K-step 32,
// double-buffered async global->LDS copies.
__launch_bounds__(256)
__global__ void k_gemm_bf16(const unsigned short* __restrict__ X,
                            const unsigned short* __restrict__ W,
                            unsigned short* __restrict__ outU16,
                            float* __restrict__ outF32,
                            int mode)
{
  __shared__ alignas(32) unsigned short lA[2][128 * 32];
  __shared__ alignas(32) unsigned short lB[2][128 * 32];

  const int tid  = threadIdx.x;
  const int lane = tid & 31, wid = tid >> 5;
  const int wm = wid & 3, wn = wid >> 2;        // 4 M-waves x 2 N-waves
  const int bm = blockIdx.x, bn = blockIdx.y;
  const int l15 = lane & 15, kh = lane >> 4;

  v8f acc[2][4];
  for (int i = 0; i < 2; i++)
    for (int j = 0; j < 4; j++)
      for (int e = 0; e < 8; e++) acc[i][j][e] = 0.f;

  const unsigned short* Abase = X + (size_t)bm * 128 * D_MODEL;  // uniform
  const unsigned short* Bbase = W + (size_t)bn * 128 * D_MODEL;  // uniform
  unsigned aB[2] = { (unsigned)(size_t)(void*)&lA[0][0],
                     (unsigned)(size_t)(void*)&lA[1][0] };
  unsigned bB[2] = { (unsigned)(size_t)(void*)&lB[0][0],
                     (unsigned)(size_t)(void*)&lB[1][0] };

  // each thread copies 2x16B for A and 2x16B for B per K-slice (8KB each tile)
  auto issue_tile = [&](int k0, int buf) {
#pragma unroll
    for (int i = 0; i < 2; i++) {
      unsigned c = (unsigned)tid + i * 256u;
      unsigned row = c >> 2, off = (c & 3u) * 16u;            // 4x16B per 64B row
      unsigned gof = (row * (unsigned)D_MODEL + (unsigned)k0) * 2u + off;
      unsigned lofA = aB[buf] + row * 64u + off;
      unsigned lofB = bB[buf] + row * 64u + off;
      asm volatile("global_load_async_to_lds_b128 %0, %1, %2"
                   :: "v"(lofA), "v"(gof), "s"(Abase) : "memory");
      asm volatile("global_load_async_to_lds_b128 %0, %1, %2"
                   :: "v"(lofB), "v"(gof), "s"(Bbase) : "memory");
    }
  };

  issue_tile(0, 0);

  const int NK = D_MODEL / 32;
  for (int kt = 0; kt < NK; kt++) {
    const int cur = kt & 1;
    if (kt + 1 < NK) {
      issue_tile((kt + 1) * 32, cur ^ 1);
      asm volatile("s_wait_asynccnt 0x4" ::: "memory");  // current slice done, next in flight
    } else {
      asm volatile("s_wait_asynccnt 0x0" ::: "memory");
    }
    __syncthreads();

    const unsigned short* la = &lA[cur][0];
    const unsigned short* lb = &lB[cur][0];
    v16bf af[2], bf[4];
#pragma unroll
    for (int am = 0; am < 2; am++) {
      int row = wm * 32 + am * 16 + l15;
      FragU f;
      f.h[0] = *(const v8us*)(la + row * 32 + kh * 8);
      f.h[1] = *(const v8us*)(la + row * 32 + 16 + kh * 8);
      af[am] = __builtin_bit_cast(v16bf, f.v);
    }
#pragma unroll
    for (int bq = 0; bq < 4; bq++) {
      int col = wn * 64 + bq * 16 + l15;
      FragU u;
      u.h[0] = *(const v8us*)(lb + col * 32 + kh * 8);
      u.h[1] = *(const v8us*)(lb + col * 32 + 16 + kh * 8);
      bf[bq] = __builtin_bit_cast(v16bf, u.v);
    }
#pragma unroll
    for (int am = 0; am < 2; am++)
#pragma unroll
      for (int bq = 0; bq < 4; bq++)
        acc[am][bq] = __builtin_amdgcn_wmma_f32_16x16x32_bf16(
            false, af[am], false, bf[bq], (short)0, acc[am][bq], false, false);
    __syncthreads();   // reads done before next issue overwrites this buffer
  }

  // ---- epilogue: uniform mode branch hoisted out of the store loops
  const int m0 = bm * 128 + wm * 32 + 8 * kh;
  const int n0 = bn * 128 + wn * 64 + l15;
  if (mode == 3) {
    for (int am = 0; am < 2; am++)
      for (int bq = 0; bq < 4; bq++)
        for (int e = 0; e < 8; e++) {
          int m = m0 + am * 16 + e, n = n0 + bq * 16;
          outF32[(size_t)m * D_MODEL + n] = acc[am][bq][e];
        }
  } else if (mode == 2) {           // V -> [b][h][d][t]
    for (int am = 0; am < 2; am++)
      for (int bq = 0; bq < 4; bq++)
        for (int e = 0; e < 8; e++) {
          int m = m0 + am * 16 + e, n = n0 + bq * 16;
          int b = m >> 11, t = m & 2047, h = n >> 7, d = n & 127;
          outU16[(((size_t)(b * N_HEADS + h)) * D_HEAD + d) * SEQ + t] =
              f2bf(acc[am][bq][e]);
        }
  } else {                          // Q/K -> [b][h][t][d]
    for (int am = 0; am < 2; am++)
      for (int bq = 0; bq < 4; bq++)
        for (int e = 0; e < 8; e++) {
          int m = m0 + am * 16 + e, n = n0 + bq * 16;
          int b = m >> 11, t = m & 2047, h = n >> 7, d = n & 127;
          outU16[(((size_t)(b * N_HEADS + h)) * SEQ + t) * D_HEAD + d] =
              f2bf(acc[am][bq][e]);
        }
  }
}

// ---------------------------------------------------------------- RoPE (in-place on bf16 Q/K)
__global__ void k_rope(unsigned short* __restrict__ Qh,
                       unsigned short* __restrict__ Kh,
                       const float* __restrict__ cosT,
                       const float* __restrict__ sinT)
{
  int i = blockIdx.x * blockDim.x + threadIdx.x;   // B*H*T*64
  unsigned short* buf = (blockIdx.y == 0) ? Qh : Kh;
  int d  = i & 63;
  int t  = (i >> 6) & 2047;
  int bh = i >> 17;
  size_t base = ((size_t)bh * SEQ + t) * D_HEAD;
  float lo = bf2f(buf[base + d]);
  float hi = bf2f(buf[base + d + 64]);
  float c0 = cosT[t * D_HEAD + d],      s0 = sinT[t * D_HEAD + d];
  float c1 = cosT[t * D_HEAD + d + 64], s1 = sinT[t * D_HEAD + d + 64];
  buf[base + d]      = f2bf(lo * c0 - hi * s0);
  buf[base + d + 64] = f2bf(hi * c1 + lo * s1);
}

// ---------------------------------------------------------------- flash attention
// grid: (SEQ/64, B*H), block 128 (4 waves x 16 queries), double-buffered K/V tiles
__launch_bounds__(128)
__global__ void k_attn(const unsigned short* __restrict__ Qh,
                       const unsigned short* __restrict__ Kh,
                       const unsigned short* __restrict__ Vt,
                       unsigned short* __restrict__ AO)
{
  __shared__ alignas(32) unsigned short lK[2][32 * 128];   // [key][d]
  __shared__ alignas(32) unsigned short lV[2][128 * 32];   // [d][key]
  __shared__ alignas(32) unsigned short lP[4][16 * 32];    // per-wave P staging

  const int tid = threadIdx.x, lane = tid & 31, wv = tid >> 5;
  const int l15 = lane & 15, kh = lane >> 4;
  const int bh = blockIdx.y;
  const int q0blk = blockIdx.x * 64;
  const int q0w = q0blk + wv * 16;

  const unsigned short* Qb = Qh + (size_t)bh * SEQ * D_HEAD;
  const unsigned short* Kb = Kh + (size_t)bh * SEQ * D_HEAD;
  const unsigned short* Vb = Vt + (size_t)bh * D_HEAD * SEQ;

  // Q fragments for this wave's 16 rows (Dh=128 -> 4 k-depth-32 frags)
  v16bf qf[4];
  {
    const unsigned short* qrow = Qb + (size_t)(q0w + l15) * D_HEAD;
#pragma unroll
    for (int kk = 0; kk < 4; kk++) {
      FragU f;
      f.h[0] = *(const v8us*)(qrow + kk * 32 + kh * 8);
      f.h[1] = *(const v8us*)(qrow + kk * 32 + 16 + kh * 8);
      qf[kk] = __builtin_bit_cast(v16bf, f.v);
    }
  }

  v8f o[8];
  for (int nt = 0; nt < 8; nt++)
    for (int e = 0; e < 8; e++) o[nt][e] = 0.f;
  float mrow[8], lrow[8];
  for (int e = 0; e < 8; e++) { mrow[e] = -1e30f; lrow[e] = 0.f; }

  unsigned kBfr[2] = { (unsigned)(size_t)(void*)&lK[0][0],
                       (unsigned)(size_t)(void*)&lK[1][0] };
  unsigned vBfr[2] = { (unsigned)(size_t)(void*)&lV[0][0],
                       (unsigned)(size_t)(void*)&lV[1][0] };
  const float scale = 0.08838834764831845f;   // 1/sqrt(128)

  // 8 async b128s per thread: K tile 32x128 (8KB) + V tile 128x32 (8KB)
  auto issue_kv = [&](int k0, int buf) {
#pragma unroll
    for (int i = 0; i < 4; i++) {
      unsigned c = (unsigned)tid + i * 128u;
      unsigned row = c >> 4, off = (c & 15u) * 16u;
      unsigned gof = ((unsigned)(k0 + row) * (unsigned)D_HEAD) * 2u + off;
      unsigned lof = kBfr[buf] + row * 256u + off;
      asm volatile("global_load_async_to_lds_b128 %0, %1, %2"
                   :: "v"(lof), "v"(gof), "s"(Kb) : "memory");
    }
#pragma unroll
    for (int i = 0; i < 4; i++) {
      unsigned c = (unsigned)tid + i * 128u;
      unsigned row = c >> 2, off = (c & 3u) * 16u;
      unsigned gof = (row * (unsigned)SEQ + (unsigned)k0) * 2u + off;
      unsigned lof = vBfr[buf] + row * 64u + off;
      asm volatile("global_load_async_to_lds_b128 %0, %1, %2"
                   :: "v"(lof), "v"(gof), "s"(Vb) : "memory");
    }
  };

  const int nkb = blockIdx.x * 2 + 2;         // key blocks of 32 up to q0blk+63
  issue_kv(0, 0);

  for (int ib = 0; ib < nkb; ib++) {
    const int k0 = ib * 32;
    const int cur = ib & 1;
    if (ib + 1 < nkb) {
      issue_kv((ib + 1) * 32, cur ^ 1);
      asm volatile("s_wait_asynccnt 0x8" ::: "memory");  // block ib landed, ib+1 in flight
    } else {
      asm volatile("s_wait_asynccnt 0x0" ::: "memory");
    }
    __syncthreads();

    if (k0 <= q0w + 15) {                      // wave-uniform causal skip
      const unsigned short* lk = &lK[cur][0];
      const unsigned short* lv = &lV[cur][0];

      // S = Q K^T : two 16x16 f32 tiles (32 keys)
      v8f st[2];
#pragma unroll
      for (int ct = 0; ct < 2; ct++) {
        v8f s; for (int e = 0; e < 8; e++) s[e] = 0.f;
        int key = ct * 16 + l15;
#pragma unroll
        for (int kk = 0; kk < 4; kk++) {
          FragU u;
          u.h[0] = *(const v8us*)(lk + key * 128 + kk * 32 + kh * 8);
          u.h[1] = *(const v8us*)(lk + key * 128 + kk * 32 + 16 + kh * 8);
          v16bf b = __builtin_bit_cast(v16bf, u.v);
          s = __builtin_amdgcn_wmma_f32_16x16x32_bf16(
              false, qf[kk], false, b, (short)0, s, false, false);
        }
        st[ct] = s;
      }

      // online softmax (rows live in lane halves; reduce over 16 lanes)
      float p0[8], p1[8];
#pragma unroll
      for (int e = 0; e < 8; e++) {
        int rowg = q0w + e + 8 * kh;
        float a  = st[0][e] * scale;
        float bb = st[1][e] * scale;
        if (k0 + l15 > rowg)      a  = -1e30f;
        if (k0 + 16 + l15 > rowg) bb = -1e30f;
        float mx = fmaxf(a, bb);
        mx = fmaxf(mx, __shfl_xor(mx, 1, 16));
        mx = fmaxf(mx, __shfl_xor(mx, 2, 16));
        mx = fmaxf(mx, __shfl_xor(mx, 4, 16));
        mx = fmaxf(mx, __shfl_xor(mx, 8, 16));
        float mnew  = fmaxf(mrow[e], mx);
        float alpha = __expf(mrow[e] - mnew);
        mrow[e] = mnew;
        float pa = __expf(a - mnew), pb = __expf(bb - mnew);
        float rsum = pa + pb;
        rsum += __shfl_xor(rsum, 1, 16);
        rsum += __shfl_xor(rsum, 2, 16);
        rsum += __shfl_xor(rsum, 4, 16);
        rsum += __shfl_xor(rsum, 8, 16);
        lrow[e] = lrow[e] * alpha + rsum;
        for (int nt = 0; nt < 8; nt++) o[nt][e] *= alpha;
        p0[e] = pa; p1[e] = pb;
      }

      // stage P (C layout) -> LDS -> reload as A fragment (16x32 bf16)
      unsigned short* Pw = &lP[wv][0];
#pragma unroll
      for (int e = 0; e < 8; e++) {
        int r = e + 8 * kh;
        Pw[r * 32 + l15]      = f2bf(p0[e]);
        Pw[r * 32 + 16 + l15] = f2bf(p1[e]);
      }
      asm volatile("s_wait_dscnt 0" ::: "memory");
      v16bf pf;
      {
        FragU f;
        f.h[0] = *(const v8us*)(Pw + l15 * 32 + kh * 8);
        f.h[1] = *(const v8us*)(Pw + l15 * 32 + 16 + kh * 8);
        pf = __builtin_bit_cast(v16bf, f.v);
      }

      // O += P V : 8 N-tiles over Dh=128
#pragma unroll
      for (int nt = 0; nt < 8; nt++) {
        int dh = nt * 16 + l15;
        FragU u;
        u.h[0] = *(const v8us*)(lv + dh * 32 + kh * 8);
        u.h[1] = *(const v8us*)(lv + dh * 32 + 16 + kh * 8);
        v16bf b = __builtin_bit_cast(v16bf, u.v);
        o[nt] = __builtin_amdgcn_wmma_f32_16x16x32_bf16(
            false, pf, false, b, (short)0, o[nt], false, false);
      }
    }
    __syncthreads();   // all reads of this buffer done before it is re-filled
  }

  // normalize and store attention output bf16 [b][t][h*128+d]
  const int b = bh >> 4, h = bh & 15;
  for (int nt = 0; nt < 8; nt++) {
    int d = nt * 16 + l15;
    for (int e = 0; e < 8; e++) {
      int t = q0w + e + 8 * kh;
      float v = o[nt][e] / lrow[e];
      AO[((size_t)b * SEQ + t) * D_MODEL + h * D_HEAD + d] = f2bf(v);
    }
  }
}

// ---------------------------------------------------------------- launch
extern "C" void kernel_launch(void* const* d_in, const int* in_sizes, int n_in,
                              void* d_out, int out_size, void* d_ws, size_t ws_size,
                              hipStream_t stream)
{
  (void)in_sizes; (void)n_in; (void)out_size; (void)ws_size;
  const float* x   = (const float*)d_in[0];
  const float* rc  = (const float*)d_in[1];
  const float* rs  = (const float*)d_in[2];
  const float* wq  = (const float*)d_in[3];
  const float* wk  = (const float*)d_in[4];
  const float* wvp = (const float*)d_in[5];
  const float* wo  = (const float*)d_in[6];
  float* out = (float*)d_out;

  const size_t NX = (size_t)MTOT * D_MODEL;     // 16,777,216 elems
  const size_t NW = (size_t)D_MODEL * D_MODEL;  //  4,194,304 elems
  unsigned short* xb  = (unsigned short*)d_ws;
  unsigned short* wqb = xb  + NX;
  unsigned short* wkb = wqb + NW;
  unsigned short* wvb = wkb + NW;
  unsigned short* wob = wvb + NW;
  unsigned short* Qh  = wob + NW;
  unsigned short* Kh  = Qh  + NX;
  unsigned short* Vt  = Kh  + NX;
  unsigned short* AO  = Vt  + NX;   // total ~192 MB of workspace

  k_f32_to_bf16<<<(int)(NX / 4 / 256), 256, 0, stream>>>(x,   xb,  (int)(NX / 4));
  k_f32_to_bf16<<<(int)(NW / 4 / 256), 256, 0, stream>>>(wq,  wqb, (int)(NW / 4));
  k_f32_to_bf16<<<(int)(NW / 4 / 256), 256, 0, stream>>>(wk,  wkb, (int)(NW / 4));
  k_f32_to_bf16<<<(int)(NW / 4 / 256), 256, 0, stream>>>(wvp, wvb, (int)(NW / 4));
  k_f32_to_bf16<<<(int)(NW / 4 / 256), 256, 0, stream>>>(wo,  wob, (int)(NW / 4));

  dim3 gg(MTOT / 128, D_MODEL / 128);
  k_gemm_bf16<<<gg, 256, 0, stream>>>(xb, wqb, Qh, nullptr, 0);
  k_gemm_bf16<<<gg, 256, 0, stream>>>(xb, wkb, Kh, nullptr, 1);
  k_gemm_bf16<<<gg, 256, 0, stream>>>(xb, wvb, Vt, nullptr, 2);

  k_rope<<<dim3(BATCH * N_HEADS * SEQ * 64 / 256, 2), 256, 0, stream>>>(Qh, Kh, rc, rs);

  k_attn<<<dim3(SEQ / 64, BATCH * N_HEADS), 128, 0, stream>>>(Qh, Kh, Vt, AO);

  k_gemm_bf16<<<gg, 256, 0, stream>>>(AO, wob, nullptr, out, 3);
}
